// HyperDHHNNConvolution_37546604102458
// MI455X (gfx1250) — matrix operand order repaired and checked by reading, not compiled, not run on hardware
//
#include <hip/hip_runtime.h>
#include <hip/hip_bf16.h>

// ---------------------------------------------------------------------------
// Problem constants (from reference): N=65536 nodes, D=256, NS=512 slots,
// HID=256, k_n=64, k_e=8. All tile dims divide evenly (no bounds checks).
// ---------------------------------------------------------------------------
#define NN 65536
#define DD 256
#define NSS 512
#define HIDD 256
#define KN 64
#define KE 8

typedef __attribute__((ext_vector_type(16))) __bf16 v16bf;
typedef __attribute__((ext_vector_type(8)))  float  v8f;

struct U256 { uint4 a, b; };

__device__ __forceinline__ unsigned short f2bf(float f) {
  unsigned int u = __builtin_bit_cast(unsigned int, f);
  u = u + 0x7FFFu + ((u >> 16) & 1u);
  return (unsigned short)(u >> 16);
}

// ---------------------------------------------------------------------------
// Generic WMMA GEMM: C[M,Nc] = act(scale * (A @ B^T) + bias)
//   A: bf16 [M,K] row-major, B: bf16 [Nc,K] row-major (so A@B^T contracts K)
//   Block = 128 threads = 4 waves; tile 64x64, K-chunk 32.
//   Each wave owns a 32x32 quadrant = 2x2 v_wmma_f32_16x16x32_bf16 tiles.
//
// Double-buffered LDS (2 x 12KB of CDNA5's 320KB/WGP): global_load_b128 for
// chunk kc+1 are issued before the WMMAs of chunk kc, stores land in the
// other buffer after the WMMAs, one barrier per iteration. All global
// addressing is pointer-increment (no per-iteration 64-bit multiplies).
// Padded LDS row stride (48 elems = 96B) keeps every fragment b128 read
// 16B-aligned while dodging bank conflicts.
// ---------------------------------------------------------------------------
__global__ __launch_bounds__(128) void gemm_bf16_wmma(
    const unsigned short* __restrict__ A, const unsigned short* __restrict__ B,
    const float* __restrict__ bias, float scale, int relu,
    float* __restrict__ Cf, unsigned short* __restrict__ Cb,
    int M, int Nc, int K)
{
  __shared__ unsigned short sA[2][64 * 48];
  __shared__ unsigned short sB[2][64 * 48];

  const int tid  = threadIdx.x;
  const int lane = tid & 31;
  const int wave = tid >> 5;
  const int wm = wave >> 1, wn = wave & 1;
  const int lh = lane >> 4, lm = lane & 15;
  const int m0 = blockIdx.y * 64;
  const int n0 = blockIdx.x * 64;

  // Per-thread staging role: rows r0 and r0+32, 8-elem segment seg.
  const int r0  = tid >> 2;   // 0..31
  const int seg = tid & 3;    // 0..3
  const int ldsOff0 = r0 * 48 + seg * 8;
  const int ldsOff1 = (r0 + 32) * 48 + seg * 8;

  const unsigned short* pA0 = A + (size_t)(m0 + r0) * K + seg * 8;
  const unsigned short* pA1 = pA0 + (size_t)32 * K;
  const unsigned short* pB0 = B + (size_t)(n0 + r0) * K + seg * 8;
  const unsigned short* pB1 = pB0 + (size_t)32 * K;

  v8f zero = {0.f, 0.f, 0.f, 0.f, 0.f, 0.f, 0.f, 0.f};
  v8f acc[2][2];
  acc[0][0] = zero; acc[0][1] = zero; acc[1][0] = zero; acc[1][1] = zero;

  // Prologue: stage chunk 0 into buffer 0.
  {
    uint4 ra0 = *reinterpret_cast<const uint4*>(pA0); pA0 += 32;
    uint4 ra1 = *reinterpret_cast<const uint4*>(pA1); pA1 += 32;
    uint4 rb0 = *reinterpret_cast<const uint4*>(pB0); pB0 += 32;
    uint4 rb1 = *reinterpret_cast<const uint4*>(pB1); pB1 += 32;
    *reinterpret_cast<uint4*>(&sA[0][ldsOff0]) = ra0;
    *reinterpret_cast<uint4*>(&sA[0][ldsOff1]) = ra1;
    *reinterpret_cast<uint4*>(&sB[0][ldsOff0]) = rb0;
    *reinterpret_cast<uint4*>(&sB[0][ldsOff1]) = rb1;
  }

  // Fragment LDS offsets (fixed across iterations).
  const int aOff[2] = { (wm * 32 + 0 * 16 + lm) * 48 + lh * 8,
                        (wm * 32 + 1 * 16 + lm) * 48 + lh * 8 };
  const int bOff[2] = { (wn * 32 + 0 * 16 + lm) * 48 + lh * 8,
                        (wn * 32 + 1 * 16 + lm) * 48 + lh * 8 };

  const int KC = K >> 5;
  for (int kc = 0; kc < KC; ++kc) {
    __syncthreads();                       // buffer `cur` is ready
    const int cur = kc & 1;

    uint4 ra0, ra1, rb0, rb1;
    const bool more = (kc + 1) < KC;
    if (more) {                            // loads for next chunk in flight
      ra0 = *reinterpret_cast<const uint4*>(pA0); pA0 += 32;
      ra1 = *reinterpret_cast<const uint4*>(pA1); pA1 += 32;
      rb0 = *reinterpret_cast<const uint4*>(pB0); pB0 += 32;
      rb1 = *reinterpret_cast<const uint4*>(pB1); pB1 += 32;
    }

    v16bf afrag[2], bfrag[2];
    // A fragment (16x32): lane holds row m=lane%16; halves of wave cover
    // K = lh*8 + [0..7] and 16 + lh*8 + [0..7].
    for (int i = 0; i < 2; ++i) {
      U256 u;
      u.a = *reinterpret_cast<const uint4*>(&sA[cur][aOff[i]]);
      u.b = *reinterpret_cast<const uint4*>(&sA[cur][aOff[i] + 16]);
      afrag[i] = __builtin_bit_cast(v16bf, u);
    }
    // B fragment (32x16): lane holds col n=lane%16; K = lh*8 + [0..7] and 16 + lh*8 + [0..7].
    for (int j = 0; j < 2; ++j) {
      U256 u;
      u.a = *reinterpret_cast<const uint4*>(&sB[cur][bOff[j]]);
      u.b = *reinterpret_cast<const uint4*>(&sB[cur][bOff[j] + 16]);
      bfrag[j] = __builtin_bit_cast(v16bf, u);
    }
    for (int i = 0; i < 2; ++i)
      for (int j = 0; j < 2; ++j)
        acc[i][j] = __builtin_amdgcn_wmma_f32_16x16x32_bf16(
            false, afrag[i], false, bfrag[j], (short)0, acc[i][j], false, false);

    if (more) {                            // stage next chunk into other buffer
      const int nxt = cur ^ 1;
      *reinterpret_cast<uint4*>(&sA[nxt][ldsOff0]) = ra0;
      *reinterpret_cast<uint4*>(&sA[nxt][ldsOff1]) = ra1;
      *reinterpret_cast<uint4*>(&sB[nxt][ldsOff0]) = rb0;
      *reinterpret_cast<uint4*>(&sB[nxt][ldsOff1]) = rb1;
    }
  }

  // Epilogue. C/D layout: lane col n=lane%16, VGPR r -> row m = r + 8*(lane/16).
  for (int i = 0; i < 2; ++i) {
    for (int j = 0; j < 2; ++j) {
      const int nn = n0 + wn * 32 + j * 16 + lm;
      const float bv = bias ? bias[nn] : 0.0f;
      for (int r = 0; r < 8; ++r) {
        const int mm = m0 + wm * 32 + i * 16 + lh * 8 + r;
        float v = acc[i][j][r] * scale + bv;
        if (relu) v = v > 0.f ? v : 0.f;
        const size_t o = (size_t)mm * Nc + nn;
        if (Cf) Cf[o] = v;
        if (Cb) Cb[o] = f2bf(v);
      }
    }
  }
}

// ---------------------------------------------------------------------------
// LayerNorm over D=256, one row per 256-thread block -> bf16 out
// ---------------------------------------------------------------------------
__global__ __launch_bounds__(256) void ln_rows(const float* __restrict__ X,
                                               const float* __restrict__ g,
                                               const float* __restrict__ b,
                                               unsigned short* __restrict__ out)
{
  __shared__ float red[256];
  const int r = blockIdx.x, t = threadIdx.x;
  const float v = X[(size_t)r * DD + t];
  red[t] = v; __syncthreads();
  for (int s = 128; s > 0; s >>= 1) { if (t < s) red[t] += red[t + s]; __syncthreads(); }
  const float mean = red[0] * (1.0f / DD); __syncthreads();
  const float d = v - mean;
  red[t] = d * d; __syncthreads();
  for (int s = 128; s > 0; s >>= 1) { if (t < s) red[t] += red[t + s]; __syncthreads(); }
  const float var = red[0] * (1.0f / DD);
  const float o = d * rsqrtf(var + 1e-5f) * g[t] + b[t];
  out[(size_t)r * DD + t] = f2bf(o);
}

// ---------------------------------------------------------------------------
// edges = mu + exp(logsigma)*noise; e = LN(edges); write e (f32), e (bf16),
// and the first half of e_cat (bf16, row stride 2D).
// ---------------------------------------------------------------------------
__global__ __launch_bounds__(256) void e_prep(const float* __restrict__ noise,
                                              const float* __restrict__ mu,
                                              const float* __restrict__ ls,
                                              const float* __restrict__ g,
                                              const float* __restrict__ bb,
                                              float* __restrict__ e_f32,
                                              unsigned short* __restrict__ e_bf,
                                              unsigned short* __restrict__ e_cat)
{
  __shared__ float red[256];
  const int r = blockIdx.x, t = threadIdx.x;
  const float v = mu[t] + expf(ls[t]) * noise[(size_t)r * DD + t];
  red[t] = v; __syncthreads();
  for (int s = 128; s > 0; s >>= 1) { if (t < s) red[t] += red[t + s]; __syncthreads(); }
  const float mean = red[0] * (1.0f / DD); __syncthreads();
  const float d = v - mean;
  red[t] = d * d; __syncthreads();
  for (int s = 128; s > 0; s >>= 1) { if (t < s) red[t] += red[t + s]; __syncthreads(); }
  const float var = red[0] * (1.0f / DD);
  const float o = d * rsqrtf(var + 1e-5f) * g[t] + bb[t];
  e_f32[(size_t)r * DD + t] = o;
  const unsigned short ob = f2bf(o);
  e_bf[(size_t)r * DD + t] = ob;
  e_cat[(size_t)r * (2 * DD) + t] = ob;
}

// ---------------------------------------------------------------------------
// Transpose + fp32->bf16: W[fin,fout] -> Wt[fout,fin]
// ---------------------------------------------------------------------------
__global__ void transpose_bf16(const float* __restrict__ src,
                               unsigned short* __restrict__ dst,
                               int fin, int fout)
{
  const int idx = blockIdx.x * 256 + threadIdx.x;
  if (idx < fin * fout) {
    const int k = idx / fout, n2 = idx - k * fout;
    dst[(size_t)n2 * fin + k] = f2bf(src[idx]);
  }
}

// ---------------------------------------------------------------------------
// Per slot row (NS=512 blocks x 1024 thr): softmax over N=65536 in a 256KB
// LDS-resident row (CDNA5 320KB-per-WG LDS), top-64 iterative argmax with
// removal, then updates[s,:] = sum_64 w_j * vv[j,:] -> second half of e_cat.
// w_j = (softmax_j + eps) / (1 + N*eps)
// ---------------------------------------------------------------------------
__global__ __launch_bounds__(1024) void topk_updates(const float* __restrict__ dots,
                                                     const float* __restrict__ vv,
                                                     unsigned short* __restrict__ e_cat)
{
  extern __shared__ unsigned char smem[];
  float* srow = (float*)smem;            // 65536 f32
  float* rv   = srow + 65536;            // 1024 f32
  int*   ri   = (int*)(rv + 1024);       // 1024 i32
  float* selv = (float*)(ri + 1024);     // 64 f32
  int*   seli = (int*)(selv + 64);       // 64 i32

  const int r = blockIdx.x, t = threadIdx.x;
  for (int i = t; i < NN; i += 1024) srow[i] = dots[(size_t)r * NN + i];
  __syncthreads();

  // row max
  float lm = -3.4e38f;
  for (int i = t; i < NN; i += 1024) lm = fmaxf(lm, srow[i]);
  rv[t] = lm; __syncthreads();
  for (int s = 512; s > 0; s >>= 1) { if (t < s) rv[t] = fmaxf(rv[t], rv[t + s]); __syncthreads(); }
  const float rmax = rv[0]; __syncthreads();

  // softmax denominator
  float lsum = 0.f;
  for (int i = t; i < NN; i += 1024) lsum += expf(srow[i] - rmax);
  rv[t] = lsum; __syncthreads();
  for (int s = 512; s > 0; s >>= 1) { if (t < s) rv[t] += rv[t + s]; __syncthreads(); }
  const float sumexp = rv[0]; __syncthreads();

  // top-64 selection with removal
  for (int it = 0; it < KN; ++it) {
    float bm = -3.4e38f; int bi = 0;
    for (int i = t; i < NN; i += 1024) { if (srow[i] > bm) { bm = srow[i]; bi = i; } }
    rv[t] = bm; ri[t] = bi; __syncthreads();
    for (int s = 512; s > 0; s >>= 1) {
      if (t < s && rv[t + s] > rv[t]) { rv[t] = rv[t + s]; ri[t] = ri[t + s]; }
      __syncthreads();
    }
    if (t == 0) { selv[it] = rv[0]; seli[it] = ri[0]; srow[ri[0]] = -3.4e38f; }
    __syncthreads();
  }

  // sparse updates = attn(top64) @ vv  ->  e_cat[:, D:2D]
  if (t < DD) {
    const float inv = 1.0f / sumexp;
    const float renorm = 1.0f / (1.0f + (float)NN * 1e-8f);
    float acc = 0.f;
    for (int s2 = 0; s2 < KN; ++s2) {
      const float w = (expf(selv[s2] - rmax) * inv + 1e-8f) * renorm;
      acc += w * vv[(size_t)seli[s2] * DD + t];
    }
    e_cat[(size_t)r * (2 * DD) + DD + t] = f2bf(acc);
  }
}

// ---------------------------------------------------------------------------
// Per node row (N=65536 blocks x 128 thr): softmax over NS=512 of dots2,
// top-8 -> H row; fused sparse x_out = x + sum_8 h_k * e_out[idx_k,:]
// ---------------------------------------------------------------------------
__global__ __launch_bounds__(128) void h_xout(const float* __restrict__ dots2,
                                              const float* __restrict__ x,
                                              const float* __restrict__ e_out,
                                              float* __restrict__ Hout,
                                              float* __restrict__ xout)
{
  __shared__ float s[NSS];
  __shared__ float hv[NSS];
  __shared__ float rv[128];
  __shared__ int   ri[128];
  __shared__ float selv[KE];
  __shared__ int   seli[KE];

  const int r = blockIdx.x, t = threadIdx.x;
  for (int i = t; i < NSS; i += 128) { s[i] = dots2[(size_t)r * NSS + i]; hv[i] = 0.f; }
  __syncthreads();

  float lm = -3.4e38f;
  for (int i = t; i < NSS; i += 128) lm = fmaxf(lm, s[i]);
  rv[t] = lm; __syncthreads();
  for (int ss = 64; ss > 0; ss >>= 1) { if (t < ss) rv[t] = fmaxf(rv[t], rv[t + ss]); __syncthreads(); }
  const float mx = rv[0]; __syncthreads();

  float lsum = 0.f;
  for (int i = t; i < NSS; i += 128) { const float e = expf(s[i] - mx); s[i] = e; lsum += e; }
  rv[t] = lsum; __syncthreads();
  for (int ss = 64; ss > 0; ss >>= 1) { if (t < ss) rv[t] += rv[t + ss]; __syncthreads(); }
  const float inv = 1.0f / rv[0]; __syncthreads();

  for (int it = 0; it < KE; ++it) {
    float bm = -1.f; int bi = 0;
    for (int i = t; i < NSS; i += 128) { if (s[i] > bm) { bm = s[i]; bi = i; } }
    rv[t] = bm; ri[t] = bi; __syncthreads();
    for (int ss = 64; ss > 0; ss >>= 1) {
      if (t < ss && rv[t + ss] > rv[t]) { rv[t] = rv[t + ss]; ri[t] = ri[t + ss]; }
      __syncthreads();
    }
    if (t == 0) {
      const float p = rv[0] * inv;
      selv[it] = p; seli[it] = ri[0];
      hv[ri[0]] = p; s[ri[0]] = -1.f;
    }
    __syncthreads();
  }

  for (int i = t; i < NSS; i += 128) Hout[(size_t)r * NSS + i] = hv[i];
  for (int c = t; c < DD; c += 128) {
    float acc = x[(size_t)r * DD + c];
    for (int k2 = 0; k2 < KE; ++k2) acc += selv[k2] * e_out[(size_t)seli[k2] * DD + c];
    xout[(size_t)r * DD + c] = acc;
  }
}

// ---------------------------------------------------------------------------
// Host-side orchestration
// ---------------------------------------------------------------------------
extern "C" void kernel_launch(void* const* d_in, const int* in_sizes, int n_in,
                              void* d_out, int out_size, void* d_ws, size_t ws_size,
                              hipStream_t stream) {
  (void)in_sizes; (void)n_in; (void)out_size; (void)ws_size;

  const float* x     = (const float*)d_in[0];
  const float* noise = (const float*)d_in[1];
  const float* e_mu  = (const float*)d_in[2];
  const float* e_ls  = (const float*)d_in[3];
  const float* ln_in_g = (const float*)d_in[4];
  const float* ln_in_b = (const float*)d_in[5];
  const float* ln_e_g  = (const float*)d_in[6];
  const float* ln_e_b  = (const float*)d_in[7];
  const float* Wq = (const float*)d_in[8];  const float* bq = (const float*)d_in[9];
  const float* Wk = (const float*)d_in[10]; const float* bk = (const float*)d_in[11];
  const float* Wv = (const float*)d_in[12]; const float* bv = (const float*)d_in[13];
  const float* W1 = (const float*)d_in[14]; const float* b1 = (const float*)d_in[15];
  const float* W2 = (const float*)d_in[16]; const float* b2 = (const float*)d_in[17];
  const float* W  = (const float*)d_in[18]; const float* b  = (const float*)d_in[19];

  float* out = (float*)d_out;
  float* out_x  = out;                                      // [N, D]
  float* out_H  = out + (size_t)NN * DD;                    // [N, NS]
  float* out_d2 = out + (size_t)NN * DD + (size_t)NN * NSS; // [N, NS]

  // Workspace carve-up (all sizes multiples of 256B)
  unsigned char* ws = (unsigned char*)d_ws;
  size_t off = 0;
  auto alloc = [&](size_t bytes) { unsigned char* p = ws + off; off += bytes; return p; };
  unsigned short* xin_bf = (unsigned short*)alloc((size_t)NN * DD * 2);
  unsigned short* kk_bf  = (unsigned short*)alloc((size_t)NN * DD * 2);
  unsigned short* q2_bf  = (unsigned short*)alloc((size_t)NN * DD * 2);
  float*          vv_f   = (float*)alloc((size_t)NN * DD * 4);
  float*          dots   = (float*)alloc((size_t)NSS * NN * 4);
  float*          e_f    = (float*)alloc((size_t)NSS * DD * 4);
  unsigned short* e_bf   = (unsigned short*)alloc((size_t)NSS * DD * 2);
  unsigned short* ecat_bf= (unsigned short*)alloc((size_t)NSS * 2 * DD * 2);
  unsigned short* q_bf   = (unsigned short*)alloc((size_t)NSS * DD * 2);
  unsigned short* h_bf   = (unsigned short*)alloc((size_t)NSS * HIDD * 2);
  unsigned short* e2_bf  = (unsigned short*)alloc((size_t)NSS * DD * 2);
  unsigned short* k2_bf  = (unsigned short*)alloc((size_t)NSS * DD * 2);
  float*          eout_f = (float*)alloc((size_t)NSS * DD * 4);
  unsigned short* WqT = (unsigned short*)alloc((size_t)DD * DD * 2);
  unsigned short* WkT = (unsigned short*)alloc((size_t)DD * DD * 2);
  unsigned short* WvT = (unsigned short*)alloc((size_t)DD * DD * 2);
  unsigned short* W1T = (unsigned short*)alloc((size_t)2 * DD * HIDD * 2);
  unsigned short* W2T = (unsigned short*)alloc((size_t)HIDD * DD * 2);
  unsigned short* WT  = (unsigned short*)alloc((size_t)DD * DD * 2);

  const float scale = 0.0625f; // D^-0.5 = 1/16

  // Weight transposes to [fout, fin] bf16
  transpose_bf16<<<(DD * DD) / 256, 256, 0, stream>>>(Wq, WqT, DD, DD);
  transpose_bf16<<<(DD * DD) / 256, 256, 0, stream>>>(Wk, WkT, DD, DD);
  transpose_bf16<<<(DD * DD) / 256, 256, 0, stream>>>(Wv, WvT, DD, DD);
  transpose_bf16<<<(2 * DD * HIDD) / 256, 256, 0, stream>>>(W1, W1T, 2 * DD, HIDD);
  transpose_bf16<<<(HIDD * DD) / 256, 256, 0, stream>>>(W2, W2T, HIDD, DD);
  transpose_bf16<<<(DD * DD) / 256, 256, 0, stream>>>(W, WT, DD, DD);

  // xin = LN(x); e chain prep
  ln_rows<<<NN, 256, 0, stream>>>(x, ln_in_g, ln_in_b, xin_bf);
  e_prep<<<NSS, 256, 0, stream>>>(noise, e_mu, e_ls, ln_e_g, ln_e_b, e_f, e_bf, ecat_bf);

  // q = relu(e @ Wq + bq)
  gemm_bf16_wmma<<<dim3(DD / 64, NSS / 64), 128, 0, stream>>>(
      e_bf, WqT, bq, 1.0f, 1, nullptr, q_bf, NSS, DD, DD);
  // kk = relu(xin @ Wk + bk), vv = relu(xin @ Wv + bv), q2 = xin @ Wq + bq
  gemm_bf16_wmma<<<dim3(DD / 64, NN / 64), 128, 0, stream>>>(
      xin_bf, WkT, bk, 1.0f, 1, nullptr, kk_bf, NN, DD, DD);
  gemm_bf16_wmma<<<dim3(DD / 64, NN / 64), 128, 0, stream>>>(
      xin_bf, WvT, bv, 1.0f, 1, vv_f, nullptr, NN, DD, DD);
  gemm_bf16_wmma<<<dim3(DD / 64, NN / 64), 128, 0, stream>>>(
      xin_bf, WqT, bq, 1.0f, 0, nullptr, q2_bf, NN, DD, DD);

  // dots = (q @ kk^T) * scale   [NS, N]
  gemm_bf16_wmma<<<dim3(NN / 64, NSS / 64), 128, 0, stream>>>(
      q_bf, kk_bf, nullptr, scale, 0, dots, nullptr, NSS, NN, DD);

  // softmax + top-64 + updates (needs 270,848 B dynamic LDS; CDNA5 WG max 320KB)
  const int topk_lds = 65536 * 4 + 1024 * 4 + 1024 * 4 + 64 * 4 + 64 * 4;
  (void)hipFuncSetAttribute((const void*)topk_updates,
                            hipFuncAttributeMaxDynamicSharedMemorySize, topk_lds);
  topk_updates<<<NSS, 1024, topk_lds, stream>>>(dots, vv_f, ecat_bf);

  // MLP: h = relu(e_cat @ W1 + b1); e2 = h @ W2 + b2
  gemm_bf16_wmma<<<dim3(HIDD / 64, NSS / 64), 128, 0, stream>>>(
      ecat_bf, W1T, b1, 1.0f, 1, nullptr, h_bf, NSS, HIDD, 2 * DD);
  gemm_bf16_wmma<<<dim3(DD / 64, NSS / 64), 128, 0, stream>>>(
      h_bf, W2T, b2, 1.0f, 0, nullptr, e2_bf, NSS, DD, HIDD);

  // k2 = relu(e2 @ Wk + bk); e_out = e2 @ W + b
  gemm_bf16_wmma<<<dim3(DD / 64, NSS / 64), 128, 0, stream>>>(
      e2_bf, WkT, bk, 1.0f, 1, nullptr, k2_bf, NSS, DD, DD);
  gemm_bf16_wmma<<<dim3(DD / 64, NSS / 64), 128, 0, stream>>>(
      e2_bf, WT, b, 1.0f, 0, eout_f, nullptr, NSS, DD, DD);

  // dots2 = (q2 @ k2^T) * scale  -> straight into d_out slice
  gemm_bf16_wmma<<<dim3(NSS / 64, NN / 64), 128, 0, stream>>>(
      q2_bf, k2_bf, nullptr, scale, 0, out_d2, nullptr, NN, NSS, DD);

  // H = topk8(softmax(dots2)); x_out = x + H @ e_out (sparse gather, fused)
  h_xout<<<NN, 128, 0, stream>>>(out_d2, x, eout_f, out_H, out_x);
}
